// RSSMoDBlock_53068615909647
// MI455X (gfx1250) — matrix operand order repaired and checked by reading, not compiled, not run on hardware
//
#include <hip/hip_runtime.h>

#define DEVI __device__ __forceinline__

constexpr int B_ = 4, S_ = 4096, D_ = 1024, H_ = 16, DH_ = 64;
constexpr int INTER_ = 4096, CAP_ = 1024;

typedef __attribute__((ext_vector_type(16))) __bf16 v16bf;
typedef __attribute__((ext_vector_type(8)))  __bf16 v8bf;
typedef __attribute__((ext_vector_type(8)))  float  v8f;

DEVI v8f wmma_bf16(v16bf a, v16bf b, v8f c) {
  return __builtin_amdgcn_wmma_f32_16x16x32_bf16(false, a, false, b, (short)0, c,
                                                 false, false);
}

DEVI v16bf cat16(v8bf lo, v8bf hi) {
  return __builtin_shufflevector(lo, hi, 0, 1, 2, 3, 4, 5, 6, 7, 8, 9, 10, 11,
                                 12, 13, 14, 15);
}

DEVI float sigmoidf_(float x) { return 1.0f / (1.0f + expf(-x)); }

// ---------------------------------------------------------------------------
// RMSNorm: one block per token (D=1024), 256 threads, one float4 per thread.
// ---------------------------------------------------------------------------
template <typename OT>
__global__ void rmsnorm_kernel(const float* __restrict__ x,
                               const float* __restrict__ w,
                               OT* __restrict__ out) {
  const int token = blockIdx.x;
  const float* xr = x + (size_t)token * D_;
  const int i4 = threadIdx.x * 4;
  float4 xv = *(const float4*)(xr + i4);
  float p = xv.x * xv.x + xv.y * xv.y + xv.z * xv.z + xv.w * xv.w;
  for (int m = 16; m >= 1; m >>= 1) p += __shfl_xor(p, m, 32);
  __shared__ float red[8];
  __shared__ float rs_sh;
  if ((threadIdx.x & 31) == 0) red[threadIdx.x >> 5] = p;
  __syncthreads();
  if (threadIdx.x == 0) {
    float t = 0.f;
    for (int i = 0; i < 8; ++i) t += red[i];
    rs_sh = rsqrtf(t / (float)D_ + 1e-6f);
  }
  __syncthreads();
  const float rs = rs_sh;
  float4 wv = *(const float4*)(w + i4);
  OT* orow = out + (size_t)token * D_ + i4;
  orow[0] = (OT)(xv.x * rs * wv.x);
  orow[1] = (OT)(xv.y * rs * wv.y);
  orow[2] = (OT)(xv.z * rs * wv.z);
  orow[3] = (OT)(xv.w * rs * wv.w);
}

// ---------------------------------------------------------------------------
// SSM diagonal scan: one thread per (b,d) channel, sequential over S.
// ---------------------------------------------------------------------------
__global__ void ssm_scan_kernel(const float* __restrict__ x,
                                const float* __restrict__ xb,
                                const float* __restrict__ a,
                                const float* __restrict__ g,
                                float* __restrict__ h) {
  const int t = blockIdx.x * 256 + threadIdx.x;
  if (t >= B_ * D_) return;
  const int b = t >> 10, d = t & (D_ - 1);
  const float dec = sigmoidf_(a[d]);
  const float om = 1.0f - dec;
  const float gain = g[d];
  size_t base = ((size_t)b * S_) * D_ + d;
  float hs = 0.f;
  for (int s = 0; s < S_; ++s) {
    size_t o = base + (size_t)s * D_;
    hs = dec * hs + om * xb[o];
    h[o] = x[o] + hs * gain;
  }
}

// ---------------------------------------------------------------------------
// Router dot: one block per token.
// ---------------------------------------------------------------------------
__global__ void router_dot_kernel(const float* __restrict__ h,
                                  const float* __restrict__ rw,
                                  float* __restrict__ scores) {
  const int token = blockIdx.x;
  const float* xr = h + (size_t)token * D_;
  const int i4 = threadIdx.x * 4;
  float4 xv = *(const float4*)(xr + i4);
  float4 wv = *(const float4*)(rw + i4);
  float p = xv.x * wv.x + xv.y * wv.y + xv.z * wv.z + xv.w * wv.w;
  for (int m = 16; m >= 1; m >>= 1) p += __shfl_xor(p, m, 32);
  __shared__ float red[8];
  if ((threadIdx.x & 31) == 0) red[threadIdx.x >> 5] = p;
  __syncthreads();
  if (threadIdx.x == 0) {
    float t = 0.f;
    for (int i = 0; i < 8; ++i) t += red[i];
    scores[token] = t;
  }
}

// Deterministic aux-loss reduction (single block).
__global__ void aux_kernel(const float* __restrict__ scores,
                           float* __restrict__ auxp) {
  float p = 0.f;
  for (int i = threadIdx.x; i < B_ * S_; i += 256) p += sigmoidf_(scores[i]);
  for (int m = 16; m >= 1; m >>= 1) p += __shfl_xor(p, m, 32);
  __shared__ float red[8];
  if ((threadIdx.x & 31) == 0) red[threadIdx.x >> 5] = p;
  __syncthreads();
  if (threadIdx.x == 0) {
    float t = 0.f;
    for (int i = 0; i < 8; ++i) t += red[i];
    *auxp = t / (float)(B_ * S_);
  }
}

// ---------------------------------------------------------------------------
// Per-batch top-CAP via bitonic sort (desc by score), then re-sort winners
// ascending by index. One block per batch, 256 threads.
// ---------------------------------------------------------------------------
__global__ void topk_kernel(const float* __restrict__ scores,
                            int* __restrict__ idx_out,
                            float* __restrict__ w_out) {
  __shared__ float ss[S_];
  __shared__ int   si[S_];
  const int b = blockIdx.x;
  const int tid = threadIdx.x;
  for (int i = tid; i < S_; i += 256) { ss[i] = scores[(size_t)b * S_ + i]; si[i] = i; }
  __syncthreads();
  for (int k = 2; k <= S_; k <<= 1) {
    for (int j = k >> 1; j > 0; j >>= 1) {
      for (int i = tid; i < S_; i += 256) {
        int ixj = i ^ j;
        if (ixj > i) {
          bool asc_pair = (((i & k) == 0) == false);  // descending overall
          bool gt = ss[i] > ss[ixj];
          if (gt == asc_pair) {
            float tf = ss[i]; ss[i] = ss[ixj]; ss[ixj] = tf;
            int ti = si[i]; si[i] = si[ixj]; si[ixj] = ti;
          }
        }
      }
      __syncthreads();
    }
  }
  for (int k = 2; k <= CAP_; k <<= 1) {
    for (int j = k >> 1; j > 0; j >>= 1) {
      for (int i = tid; i < CAP_; i += 256) {
        int ixj = i ^ j;
        if (ixj > i) {
          bool asc_pair = (((i & k) == 0) == true);   // ascending by index
          bool gt = si[i] > si[ixj];
          if (gt == asc_pair) {
            int ti = si[i]; si[i] = si[ixj]; si[ixj] = ti;
            float tf = ss[i]; ss[i] = ss[ixj]; ss[ixj] = tf;
          }
        }
      }
      __syncthreads();
    }
  }
  for (int c = tid; c < CAP_; c += 256) {
    idx_out[b * CAP_ + c] = si[c];
    w_out[b * CAP_ + c] = sigmoidf_(ss[c]);
  }
}

// ---------------------------------------------------------------------------
// Gather selected tokens + rmsnorm -> bf16. One block per (b,c).
// ---------------------------------------------------------------------------
__global__ void gather_norm_kernel(const float* __restrict__ h,
                                   const int* __restrict__ idx,
                                   const float* __restrict__ w,
                                   __bf16* __restrict__ xn) {
  const int bc = blockIdx.x;
  const int b = bc / CAP_;
  const int tok = idx[bc];
  const float* xr = h + ((size_t)b * S_ + tok) * D_;
  const int i4 = threadIdx.x * 4;
  float4 xv = *(const float4*)(xr + i4);
  float p = xv.x * xv.x + xv.y * xv.y + xv.z * xv.z + xv.w * xv.w;
  for (int m = 16; m >= 1; m >>= 1) p += __shfl_xor(p, m, 32);
  __shared__ float red[8];
  __shared__ float rs_sh;
  if ((threadIdx.x & 31) == 0) red[threadIdx.x >> 5] = p;
  __syncthreads();
  if (threadIdx.x == 0) {
    float t = 0.f;
    for (int i = 0; i < 8; ++i) t += red[i];
    rs_sh = rsqrtf(t / (float)D_ + 1e-6f);
  }
  __syncthreads();
  const float rs = rs_sh;
  float4 wv = *(const float4*)(w + i4);
  __bf16* orow = xn + (size_t)bc * D_ + i4;
  orow[0] = (__bf16)(xv.x * rs * wv.x);
  orow[1] = (__bf16)(xv.y * rs * wv.y);
  orow[2] = (__bf16)(xv.z * rs * wv.z);
  orow[3] = (__bf16)(xv.w * rs * wv.w);
}

// ---------------------------------------------------------------------------
// WMMA GEMM: C = A(bf16, MxK, row-major) * Bm(f32, KxN).
// Block tile 256x64, 8 waves stacked in M, wave tile 32x64 (2x4 16x16x32
// WMMA) -> 8 WMMA per (4 global b128 + 8 ds b128) per K-step.
// A fragments are two contiguous 16B chunks per the ISA 16-bit A layout;
// B staged transposed in LDS so each B fragment is two ds_load_b128.
// EPI: 0=f32, 1=bf16*scale, 2=silu->bf16, 3=f32 accumulate, 4=bf16 transposed
//      ([b][col][cap] for the V buffer).
// ---------------------------------------------------------------------------
template <int EPI>
__global__ void gemm_bf16_kernel(const __bf16* __restrict__ A, int lda,
                                 const float* __restrict__ Bm, int ldb,
                                 void* __restrict__ Cp, int ldc,
                                 int M, int N, int K, float scale) {
  __shared__ __bf16 BsT[64][40];  // [col][k], rows 80B apart (16B aligned)
  const int tid = threadIdx.x;
  const int lane = tid & 31;
  const int wm = tid >> 5;  // 0..7, wave's 32-row stripe
  const int l15 = lane & 15;
  const int half = lane >> 4;
  const int m_blk = blockIdx.y * 256;
  const int n_blk = blockIdx.x * 64;

  v8f acc[2][4];
  for (int i = 0; i < 2; ++i)
    for (int j = 0; j < 4; ++j)
      for (int r = 0; r < 8; ++r) acc[i][j][r] = 0.f;

  const __bf16* arow[2];
  arow[0] = A + (size_t)(m_blk + wm * 32 + l15) * lda;
  arow[1] = A + (size_t)(m_blk + wm * 32 + 16 + l15) * lda;

  for (int k0 = 0; k0 < K; k0 += 32) {
    for (int t = tid; t < 32 * 64; t += 256) {
      int r = t >> 6, c = t & 63;
      BsT[c][r] = (__bf16)Bm[(size_t)(k0 + r) * ldb + (n_blk + c)];
    }
    __syncthreads();

    v16bf af[2], bfr[4];
#pragma unroll
    for (int rt = 0; rt < 2; ++rt) {
      const __bf16* rp = arow[rt] + k0;
      v8bf lo = *(const v8bf*)(rp + half * 8);
      v8bf hi = *(const v8bf*)(rp + 16 + half * 8);
      af[rt] = cat16(lo, hi);
    }
#pragma unroll
    for (int ct = 0; ct < 4; ++ct) {
      int col = ct * 16 + l15;
      v8bf lo = *(const v8bf*)&BsT[col][half * 8];
      v8bf hi = *(const v8bf*)&BsT[col][16 + half * 8];
      bfr[ct] = cat16(lo, hi);
    }
#pragma unroll
    for (int rt = 0; rt < 2; ++rt)
#pragma unroll
      for (int ct = 0; ct < 4; ++ct)
        acc[rt][ct] = wmma_bf16(af[rt], bfr[ct], acc[rt][ct]);
    __syncthreads();
  }

#pragma unroll
  for (int rt = 0; rt < 2; ++rt)
#pragma unroll
    for (int ct = 0; ct < 4; ++ct)
#pragma unroll
      for (int r = 0; r < 8; ++r) {
        int row = m_blk + wm * 32 + rt * 16 + r + (half ? 8 : 0);
        int col = n_blk + ct * 16 + l15;
        float v = acc[rt][ct][r];
        if (EPI == 0) {
          ((float*)Cp)[(size_t)row * ldc + col] = v;
        } else if (EPI == 1) {
          ((__bf16*)Cp)[(size_t)row * ldc + col] = (__bf16)(v * scale);
        } else if (EPI == 2) {
          ((__bf16*)Cp)[(size_t)row * ldc + col] = (__bf16)(v * sigmoidf_(v));
        } else if (EPI == 3) {
          ((float*)Cp)[(size_t)row * ldc + col] += v;
        } else {  // 4: transposed bf16 store for V: [b][col][cap]
          int b = row >> 10, c = row & (CAP_ - 1);
          ((__bf16*)Cp)[((size_t)b * D_ + col) * CAP_ + c] = (__bf16)v;
        }
      }
}

// ---------------------------------------------------------------------------
// Flash-style WMMA causal attention over the compressed CAP tokens.
// One wave per 16-query tile of one (b,h); 4 independent waves / block.
// Q pre-scaled by 1/sqrt(DH); row sums via WMMA against a ones fragment;
// V pre-transposed: vT[b][h*64+dh][cap]. Row-max reduction is stage-major so
// 8 ds_bpermute issue per s_wait_dscnt instead of 1.
// ---------------------------------------------------------------------------
__global__ void attn_kernel(const __bf16* __restrict__ qb,
                            const __bf16* __restrict__ kb,
                            const __bf16* __restrict__ vT,
                            __bf16* __restrict__ ob) {
  const int lane = threadIdx.x & 31;
  const int wv = threadIdx.x >> 5;  // 0..3
  const int gw = blockIdx.x * 4 + wv;
  const int qt = gw & 63;
  const int hh = (gw >> 6) & 15;
  const int b = gw >> 10;
  const int qbase = qt * 16;
  const int l15 = lane & 15;
  const int half = lane >> 4;

  __shared__ __bf16 pst[4][16][40];  // per-wave P staging, rows 80B (aligned)

  const size_t hoff = ((size_t)b * CAP_) * D_ + hh * DH_;

  v16bf qf[2];
  {
    const __bf16* qrow = qb + hoff + (size_t)(qbase + l15) * D_;
#pragma unroll
    for (int f = 0; f < 2; ++f) {
      v8bf lo = *(const v8bf*)(qrow + f * 32 + half * 8);
      v8bf hi = *(const v8bf*)(qrow + f * 32 + 16 + half * 8);
      qf[f] = cat16(lo, hi);
    }
  }

  v16bf onesf;
#pragma unroll
  for (int e = 0; e < 16; ++e) onesf[e] = (__bf16)1.0f;

  float m_[8];
  v8f O[4], L;
  for (int r = 0; r < 8; ++r) { m_[r] = -1e30f; L[r] = 0.f; }
  for (int nt = 0; nt < 4; ++nt)
    for (int r = 0; r < 8; ++r) O[nt][r] = 0.f;

  for (int kb0 = 0; kb0 <= qbase; kb0 += 32) {
    v8f sA, sB;
    for (int r = 0; r < 8; ++r) { sA[r] = 0.f; sB[r] = 0.f; }
#pragma unroll
    for (int hk = 0; hk < 2; ++hk) {
      const __bf16* krow = kb + hoff + (size_t)(kb0 + hk * 16 + l15) * D_;
      v16bf kf0, kf1;
      {
        v8bf lo = *(const v8bf*)(krow + half * 8);
        v8bf hi = *(const v8bf*)(krow + 16 + half * 8);
        kf0 = cat16(lo, hi);
      }
      {
        v8bf lo = *(const v8bf*)(krow + 32 + half * 8);
        v8bf hi = *(const v8bf*)(krow + 48 + half * 8);
        kf1 = cat16(lo, hi);
      }
      if (hk == 0) {
        sA = wmma_bf16(qf[0], kf0, sA);
        sA = wmma_bf16(qf[1], kf1, sA);
      } else {
        sB = wmma_bf16(qf[0], kf0, sB);
        sB = wmma_bf16(qf[1], kf1, sB);
      }
    }
    // mask + row max (stage-major: batch 8 independent shuffles per step)
    v8f rm;
#pragma unroll
    for (int r = 0; r < 8; ++r) {
      int rowq = qbase + r + half * 8;
      float a = sA[r];
      float bv = sB[r];
      if (kb0 + l15 > rowq) a = -1e30f;
      if (kb0 + 16 + l15 > rowq) bv = -1e30f;
      sA[r] = a;
      sB[r] = bv;
      rm[r] = fmaxf(a, bv);
    }
#pragma unroll
    for (int msk = 1; msk < 16; msk <<= 1) {
#pragma unroll
      for (int r = 0; r < 8; ++r)
        rm[r] = fmaxf(rm[r], __shfl_xor(rm[r], msk, 32));
    }
#pragma unroll
    for (int r = 0; r < 8; ++r) {
      float nm = fmaxf(m_[r], rm[r]);
      float alpha = expf(m_[r] - nm);
      m_[r] = nm;
      pst[wv][r + half * 8][l15] = (__bf16)expf(sA[r] - nm);
      pst[wv][r + half * 8][16 + l15] = (__bf16)expf(sB[r] - nm);
      L[r] *= alpha;
      for (int nt = 0; nt < 4; ++nt) O[nt][r] *= alpha;
    }
    asm volatile("s_wait_dscnt 0" ::: "memory");
    v16bf pf;
    {
      const __bf16* prow = &pst[wv][l15][0];
      v8bf lo = *(const v8bf*)(prow + half * 8);
      v8bf hi = *(const v8bf*)(prow + 16 + half * 8);
      pf = cat16(lo, hi);
    }
    L = wmma_bf16(pf, onesf, L);  // row sums (replicated across cols)
#pragma unroll
    for (int nt = 0; nt < 4; ++nt) {
      const __bf16* vrow =
          vT + ((size_t)b * D_ + hh * DH_ + nt * 16 + l15) * CAP_ + kb0;
      v8bf lo = *(const v8bf*)(vrow + half * 8);
      v8bf hi = *(const v8bf*)(vrow + 16 + half * 8);
      v16bf vf = cat16(lo, hi);
      O[nt] = wmma_bf16(pf, vf, O[nt]);
    }
  }

#pragma unroll
  for (int nt = 0; nt < 4; ++nt)
#pragma unroll
    for (int r = 0; r < 8; ++r) {
      int row = qbase + r + half * 8;
      ob[hoff + (size_t)row * D_ + nt * 16 + l15] = (__bf16)(O[nt][r] / L[r]);
    }
}

// ---------------------------------------------------------------------------
// Weighted scatter-add of attn_out back to h (idx unique per batch).
// ---------------------------------------------------------------------------
__global__ void scatter_kernel(const float* __restrict__ attn_out,
                               const int* __restrict__ idx,
                               const float* __restrict__ wts,
                               float* __restrict__ h) {
  const int bc = blockIdx.x;
  const int b = bc / CAP_;
  const int tok = idx[bc];
  const float w = wts[bc];
  float* hr = h + ((size_t)b * S_ + tok) * D_ + threadIdx.x * 4;
  const float* ar = attn_out + (size_t)bc * D_ + threadIdx.x * 4;
  float4 av = *(const float4*)ar;
  float4 hv = *(const float4*)hr;
  hv.x += av.x * w;
  hv.y += av.y * w;
  hv.z += av.z * w;
  hv.w += av.w * w;
  *(float4*)hr = hv;
}

// ---------------------------------------------------------------------------
extern "C" void kernel_launch(void* const* d_in, const int* in_sizes, int n_in,
                              void* d_out, int out_size, void* d_ws, size_t ws_size,
                              hipStream_t stream) {
  const float* hidden = (const float*)d_in[0];
  const float* ssm_norm_w = (const float*)d_in[1];
  const float* ssm_a = (const float*)d_in[2];
  const float* ssm_gain = (const float*)d_in[3];
  const float* router_w = (const float*)d_in[4];
  const float* attn_norm_w = (const float*)d_in[5];
  const float* wq = (const float*)d_in[6];
  const float* wk = (const float*)d_in[7];
  const float* wv = (const float*)d_in[8];
  const float* wo = (const float*)d_in[9];
  const float* ffn_norm_w = (const float*)d_in[10];
  const float* w1 = (const float*)d_in[11];
  const float* w2 = (const float*)d_in[12];

  float* h = (float*)d_out;                // [B,S,D] output
  float* auxp = h + (size_t)B_ * S_ * D_;  // aux_loss scalar

  char* p = (char*)d_ws;
  auto carve = [&](size_t bytes) {
    void* r = (void*)p;
    p += (bytes + 255) & ~(size_t)255;
    return r;
  };
  float*  xb       = (float*)carve(sizeof(float) * B_ * S_ * D_);
  float*  scores   = (float*)carve(sizeof(float) * B_ * S_);
  int*    idx      = (int*)carve(sizeof(int) * B_ * CAP_);
  float*  wts      = (float*)carve(sizeof(float) * B_ * CAP_);
  __bf16* xn       = (__bf16*)carve(sizeof(__bf16) * B_ * CAP_ * D_);
  __bf16* qbuf     = (__bf16*)carve(sizeof(__bf16) * B_ * CAP_ * D_);
  __bf16* kbuf     = (__bf16*)carve(sizeof(__bf16) * B_ * CAP_ * D_);
  __bf16* vTbuf    = (__bf16*)carve(sizeof(__bf16) * B_ * CAP_ * D_);
  __bf16* obuf     = (__bf16*)carve(sizeof(__bf16) * B_ * CAP_ * D_);
  float*  attn_out = (float*)carve(sizeof(float) * B_ * CAP_ * D_);
  __bf16* xf       = (__bf16*)carve(sizeof(__bf16) * B_ * S_ * D_);
  __bf16* g1       = (__bf16*)carve(sizeof(__bf16) * B_ * S_ * 1024);

  const float kInvSqrtDH = 0.125f;  // 1/sqrt(64)

  // 1. SSM pre-norm, scan, residual (h lives in d_out)
  rmsnorm_kernel<float><<<B_ * S_, 256, 0, stream>>>(hidden, ssm_norm_w, xb);
  ssm_scan_kernel<<<(B_ * D_ + 255) / 256, 256, 0, stream>>>(hidden, xb, ssm_a,
                                                             ssm_gain, h);
  // 2. Router + aux + top-K
  router_dot_kernel<<<B_ * S_, 256, 0, stream>>>(h, router_w, scores);
  aux_kernel<<<1, 256, 0, stream>>>(scores, auxp);
  topk_kernel<<<B_, 256, 0, stream>>>(scores, idx, wts);
  // 3. Gather + attn pre-norm (bf16)
  gather_norm_kernel<<<B_ * CAP_, 256, 0, stream>>>(h, idx, attn_norm_w, xn);
  // 4. QKV projections (WMMA); Q pre-scaled, V stored transposed
  {
    dim3 g(D_ / 64, (B_ * CAP_) / 256);
    gemm_bf16_kernel<1><<<g, 256, 0, stream>>>(xn, D_, wq, D_, (void*)qbuf, D_,
                                               B_ * CAP_, D_, D_, kInvSqrtDH);
    gemm_bf16_kernel<1><<<g, 256, 0, stream>>>(xn, D_, wk, D_, (void*)kbuf, D_,
                                               B_ * CAP_, D_, D_, 1.0f);
    gemm_bf16_kernel<4><<<g, 256, 0, stream>>>(xn, D_, wv, D_, (void*)vTbuf, D_,
                                               B_ * CAP_, D_, D_, 1.0f);
  }
  // 5. Causal attention over compressed tokens (WMMA flash)
  attn_kernel<<<(B_ * H_ * (CAP_ / 16)) / 4, 128, 0, stream>>>(qbuf, kbuf,
                                                               vTbuf, obuf);
  // 6. Output projection (WMMA)
  {
    dim3 g(D_ / 64, (B_ * CAP_) / 256);
    gemm_bf16_kernel<0><<<g, 256, 0, stream>>>(obuf, D_, wo, D_,
                                               (void*)attn_out, D_, B_ * CAP_,
                                               D_, D_, 1.0f);
  }
  // 7. Weighted scatter-add back onto residual stream
  scatter_kernel<<<B_ * CAP_, 256, 0, stream>>>(attn_out, idx, wts, h);
  // 8. FFN: rmsnorm -> silu(x@w1) @ w2, chunked over INTER in 4 passes
  rmsnorm_kernel<__bf16><<<B_ * S_, 256, 0, stream>>>(h, ffn_norm_w, xf);
  for (int nc = 0; nc < INTER_ / 1024; ++nc) {
    dim3 g1g(1024 / 64, (B_ * S_) / 256);
    gemm_bf16_kernel<2><<<g1g, 256, 0, stream>>>(xf, D_, w1 + nc * 1024, INTER_,
                                                 (void*)g1, 1024, B_ * S_, 1024,
                                                 D_, 1.0f);
    gemm_bf16_kernel<3><<<g1g, 256, 0, stream>>>(
        g1, 1024, w2 + (size_t)nc * 1024 * D_, D_, (void*)h, D_, B_ * S_, D_,
        1024, 1.0f);
  }
  (void)in_sizes; (void)n_in; (void)out_size; (void)ws_size;
}
